// OPFModel_84808424227066
// MI455X (gfx1250) — compile-verified, matrix operand order checked by reading, not run on hardware
//
#include <hip/hip_runtime.h>

typedef _Float16 v16h __attribute__((ext_vector_type(16)));
typedef _Float16 v8h  __attribute__((ext_vector_type(8)));
typedef float    v8f  __attribute__((ext_vector_type(8)));
typedef float    v4f  __attribute__((ext_vector_type(4)));

#define HIDDEN 128
#define K2H    256                         // concat dim = 2*HIDDEN
#define LDS_PAD 8                          // halfs; 16B shift per row -> conflict-free b128 reads
#define LDS_STRIDE (K2H + LDS_PAD)         // halfs per LDS row
#define WAVES_PER_BLOCK 8
#define EDGES_PER_WAVE 32                  // 2 M-tiles of 16
#define EDGES_PER_BLOCK (WAVES_PER_BLOCK * EDGES_PER_WAVE)

extern "C" __global__ __launch_bounds__(256)
void opf_edge_head_kernel(const float* __restrict__ node_emb,
                          const long long* __restrict__ edge_index,
                          const float* __restrict__ W1,
                          const float* __restrict__ b1,
                          const float* __restrict__ W2,
                          const float* __restrict__ b2,
                          float* __restrict__ out,
                          int n_edges)
{
    extern __shared__ _Float16 lds_w[];    // [HIDDEN][LDS_STRIDE] == W1^T in f16

    const int tid = threadIdx.x;

    // ---- Stage W1^T into LDS as f16 (one-time per block, coalesced global reads) ----
    // W1 is [K2H][HIDDEN] row-major; lds_w[n][k] = (f16)W1[k][n]
    for (int i = tid; i < K2H * HIDDEN; i += 256) {
        const int k = i >> 7;       // / HIDDEN
        const int n = i & (HIDDEN - 1);
        lds_w[n * LDS_STRIDE + k] = (_Float16)W1[i];
    }
    __syncthreads();

    const int wave = tid >> 5;
    const int lane = tid & 31;
    const int mlo  = lane & 15;    // A row-in-tile / B column-in-tile
    const int hi   = lane >> 4;    // K-half selector per WMMA fragment layout

    const long long tileBase =
        (long long)blockIdx.x * EDGES_PER_BLOCK + (long long)wave * EDGES_PER_WAVE;

    // Each lane owns two edges (M-tile 0: rows 0..15, M-tile 1: rows 16..31)
    const float* srcPtr[2];
    const float* dstPtr[2];
    #pragma unroll
    for (int mi = 0; mi < 2; ++mi) {
        long long e = tileBase + mi * 16 + mlo;
        if (e >= (long long)n_edges) e = (long long)n_edges - 1;
        const long long s = edge_index[e];
        const long long d = edge_index[(long long)n_edges + e];
        srcPtr[mi] = node_emb + s * HIDDEN;
        dstPtr[mi] = node_emb + d * HIDDEN;
    }

    v8f acc[2][8] = {};   // [M-tile][N-tile], fp32 accumulators

    // ---- Prologue: issue raw f32 gather for K-step 0 ----
    v4f raw[2][4];
    #pragma unroll
    for (int mi = 0; mi < 2; ++mi) {
        const float* p0 = srcPtr[mi] + hi * 8;     // K = hi*8 .. +7
        raw[mi][0] = *(const v4f*)(p0);
        raw[mi][1] = *(const v4f*)(p0 + 4);
        raw[mi][2] = *(const v4f*)(p0 + 16);       // K = 16+hi*8 .. +7
        raw[mi][3] = *(const v4f*)(p0 + 20);
    }

    #pragma unroll
    for (int kk = 0; kk < K2H; kk += 32) {
        // ---- convert this K-step's gathered f32 -> f16 A fragments ----
        v16h a[2];
        #pragma unroll
        for (int mi = 0; mi < 2; ++mi) {
            #pragma unroll
            for (int q = 0; q < 4; ++q) {
                #pragma unroll
                for (int j = 0; j < 4; ++j)
                    a[mi][q * 4 + j] = (_Float16)raw[mi][q][j];
            }
        }

        // ---- issue NEXT K-step's global gather early (hidden under WMMAs) ----
        v4f rawn[2][4];
        if (kk + 32 < K2H) {
            const int  kn  = kk + 32;
            const int  kbn = kn & (HIDDEN - 1);
            #pragma unroll
            for (int mi = 0; mi < 2; ++mi) {
                const float* row = (kn < HIDDEN) ? srcPtr[mi] : dstPtr[mi];
                const float* p0  = row + kbn + hi * 8;
                rawn[mi][0] = *(const v4f*)(p0);
                rawn[mi][1] = *(const v4f*)(p0 + 4);
                rawn[mi][2] = *(const v4f*)(p0 + 16);
                rawn[mi][3] = *(const v4f*)(p0 + 20);
            }
        }

        // ---- preload ALL 8 B fragments for this K-step (one deep DS clause) ----
        v16h bf[8];
        #pragma unroll
        for (int t = 0; t < 8; ++t) {
            const _Float16* wrow = lds_w + (t * 16 + mlo) * LDS_STRIDE + kk + hi * 8;
            v8h blo = *(const v8h*)(wrow);        // K = kk+hi*8 .. +7
            v8h bhi = *(const v8h*)(wrow + 16);   // K = kk+16+hi*8 .. +7
            bf[t] = __builtin_shufflevector(blo, bhi,
                       0,1,2,3,4,5,6,7,8,9,10,11,12,13,14,15);
        }

        // ---- 16 back-to-back WMMAs (B reused across both M-tiles) ----
        #pragma unroll
        for (int t = 0; t < 8; ++t) {
            acc[0][t] = __builtin_amdgcn_wmma_f32_16x16x32_f16(
                            false, a[0], false, bf[t], (short)0, acc[0][t], false, false);
            acc[1][t] = __builtin_amdgcn_wmma_f32_16x16x32_f16(
                            false, a[1], false, bf[t], (short)0, acc[1][t], false, false);
        }

        // rotate double buffer (renamed away by full unroll)
        #pragma unroll
        for (int mi = 0; mi < 2; ++mi)
            #pragma unroll
            for (int q = 0; q < 4; ++q)
                raw[mi][q] = rawn[mi][q];
    }

    // ---- Epilogue: +b1, ReLU, h @ W2 + b2, half-wave shuffle reduction ----
    // Lane holds h[m][n] with m = r + 8*hi (per M-tile), n = t*16 + mlo
    float bb1[8], w20[8], w21[8];
    #pragma unroll
    for (int t = 0; t < 8; ++t) {
        const int n = t * 16 + mlo;
        bb1[t] = b1[n];
        w20[t] = W2[n * 2 + 0];
        w21[t] = W2[n * 2 + 1];
    }
    const float bias0 = b2[0];
    const float bias1 = b2[1];

    #pragma unroll
    for (int mi = 0; mi < 2; ++mi) {
        #pragma unroll
        for (int r = 0; r < 8; ++r) {
            float s0 = 0.f, s1 = 0.f;
            #pragma unroll
            for (int t = 0; t < 8; ++t) {
                float h = acc[mi][t][r] + bb1[t];
                h = h > 0.f ? h : 0.f;
                s0 += h * w20[t];
                s1 += h * w21[t];
            }
            // reduce over the 16 lanes of this half-wave (xor masks 8,4,2,1)
            #pragma unroll
            for (int m = 8; m >= 1; m >>= 1) {
                s0 += __shfl_xor(s0, m, 32);
                s1 += __shfl_xor(s1, m, 32);
            }
            if (mlo == 0) {
                const long long eo = tileBase + mi * 16 + r + 8 * hi;
                if (eo < (long long)n_edges) {
                    out[eo * 2 + 0] = s0 + bias0;
                    out[eo * 2 + 1] = s1 + bias1;
                }
            }
        }
    }
}

extern "C" void kernel_launch(void* const* d_in, const int* in_sizes, int n_in,
                              void* d_out, int out_size, void* d_ws, size_t ws_size,
                              hipStream_t stream) {
    (void)n_in; (void)out_size; (void)d_ws; (void)ws_size;
    const float*     node_emb   = (const float*)d_in[0];
    const long long* edge_index = (const long long*)d_in[1];  // int64 [2, E]
    const float*     W1         = (const float*)d_in[2];
    const float*     b1         = (const float*)d_in[3];
    const float*     W2         = (const float*)d_in[4];
    const float*     b2         = (const float*)d_in[5];
    float*           out        = (float*)d_out;

    const int n_edges = in_sizes[1] / 2;
    const int blocks  = (n_edges + EDGES_PER_BLOCK - 1) / EDGES_PER_BLOCK;
    const size_t shmem = (size_t)HIDDEN * LDS_STRIDE * sizeof(_Float16);  // 67.6 KB

    opf_edge_head_kernel<<<blocks, 256, shmem, stream>>>(
        node_emb, edge_index, W1, b1, W2, b2, out, n_edges);
}
